// NonLocalMeansFast_53764400611632
// MI455X (gfx1250) — matrix operand order repaired and checked by reading, not compiled, not run on hardware
//
#include <hip/hip_runtime.h>

typedef __attribute__((ext_vector_type(16))) _Float16 v16h;
typedef __attribute__((ext_vector_type(8)))  float    v8f;

#define IMG_H 512
#define IMG_W 512
#define HW (IMG_H * IMG_W)
#define SR 5                 // search radius (11x11 window)
#define PR 2                 // patch radius (5x5 box)
#define YH 30                // 16 + 2*(SR+PR) luma halo region
#define RH 26                // 16 + 2*SR rgb halo region
#define NW 4                 // waves per block
#define NSHIFT 121

// Non-local means, one 16x16 output tile per block, 4 wave32s split the 121
// search shifts.  The 5x5 box filter of squared luma differences runs on the
// matrix pipe (3x v_wmma_f32_16x16x32_f16 per shift with banded one-matrices);
// weights use raw v_sqrt_f32 / v_exp_f32 so the transcendental work co-executes
// with the WMMAs.  dist^2 is a sum of non-negative products (band-of-ones x
// squares, RNE f32 accumulation) so no clamp is needed before the sqrt.
__global__ __launch_bounds__(32 * NW)
void nlm_wmma_kernel(const float* __restrict__ rgb,
                     const float* __restrict__ sigma,
                     float* __restrict__ out)
{
    __shared__ float ylds[YH * YH];          // luma, origin tile-(7,7), wrapped
    __shared__ float rgbl[3][RH * RH];       // rgb,  origin tile-(5,5), wrapped
    __shared__ float vlds[NW][16 * 20];      // per-wave V relayout scratch
    __shared__ float part[NW][4][256];       // per-wave partial sums (r,g,b,w)

    const int tid   = threadIdx.x;
    const int lane  = tid & 31;
    const int wave  = tid >> 5;
    const int Mrow  = lane & 15;             // N for B/C/D layout, M for A layout
    const int khalf = lane >> 4;             // 0: lanes 0-15, 1: lanes 16-31
    const int t0y   = blockIdx.y * 16;
    const int t0x   = blockIdx.x * 16;

    __builtin_prefetch(&rgb[t0y * IMG_W + t0x], 0, 0);   // global_prefetch_b8

    // ---- cooperative staging of halo regions (circular wrap, H=W=512) ----
    for (int idx = tid; idx < YH * YH; idx += 32 * NW) {
        const int i  = idx / YH, j = idx - i * YH;
        const int gy = (t0y + i - (SR + PR)) & (IMG_H - 1);
        const int gx = (t0x + j - (SR + PR)) & (IMG_W - 1);
        const float r = rgb[0 * HW + gy * IMG_W + gx];
        const float g = rgb[1 * HW + gy * IMG_W + gx];
        const float b = rgb[2 * HW + gy * IMG_W + gx];
        ylds[idx] = 0.299f * r + 0.587f * g + 0.114f * b;
    }
    for (int idx = tid; idx < RH * RH; idx += 32 * NW) {
        const int i  = idx / RH, j = idx - i * RH;
        const int gy = (t0y + i - SR) & (IMG_H - 1);
        const int gx = (t0x + j - SR) & (IMG_W - 1);
        rgbl[0][idx] = rgb[0 * HW + gy * IMG_W + gx];
        rgbl[1][idx] = rgb[1 * HW + gy * IMG_W + gx];
        rgbl[2][idx] = rgb[2 * HW + gy * IMG_W + gx];
    }
    __syncthreads();

    const float hh      = fmaxf(sigma[0] * 2.0f, 0.0f) + 1e-6f;
    // w = exp(-dist/h) = exp2(dist * (-log2(e)/h)); fold the scale into one constant
    const float nlog2h  = -1.44269504088896340736f / hh;

    // ---- constant banded one-matrices in WMMA register layouts ----
    // Av (A layout, 16x32): Av[M][K] = 1 iff M <= K <= M+4   (vertical sum)
    // Bh (B layout, 32x16): Bh[K][N] = 1 iff N <= K <= N+4   (horizontal sum)
    v16h avband, bhband;
#pragma unroll
    for (int e = 0; e < 16; ++e) {
        const int v  = e >> 1, p = e & 1;
        const int ka = ((v & 3) << 1) + p + ((v >> 2) << 4) + (khalf << 3);
        avband[e] = (_Float16)((ka >= Mrow && ka <= Mrow + 4 && ka < 20) ? 1.0f : 0.0f);
        const int kb = ((v & 3) << 1) + p + ((v >> 2) << 4) + (khalf << 3);
        bhband[e] = (_Float16)((kb >= Mrow && kb <= Mrow + 4) ? 1.0f : 0.0f);
    }

    v8f accr = {}, accg = {}, accb = {}, accw = {};
    float* myv = vlds[wave];

    // ---- search-shift loop: each wave owns shifts k = wave, wave+4, ... ----
    for (int k = wave; k < NSHIFT; k += NW) {
        const int dy = (k % 11) - SR;
        const int dx = (k / 11) - SR;

        // Stage 1: V[m][j] = sum_{t=0..4} S[m+t][j], j in 0..19, as two WMMAs.
        // S[a][b] = (y(p) - y(p - (dy,dx)))^2 at p = tile0 + (a-2, b-2),
        // built directly from ylds into the B fragment.
        v8f d1, d2;
#pragma unroll
        for (int half = 0; half < 2; ++half) {
            const int off = half * 4;                      // column offset 0 / 4
            v16h bfrag;
#pragma unroll
            for (int e = 0; e < 16; ++e) {
                const int v  = e >> 1, p = e & 1;
                const int kb = ((v & 3) << 1) + p + ((v >> 2) << 4) + (khalf << 3); // S row 0..31
                float val = 0.0f;
                if (kb < 20) {
                    const int jj = Mrow + off;              // S col 0..19
                    const float y0 = ylds[(kb + SR) * YH + (jj + SR)];
                    const float y1 = ylds[(kb + SR - dy) * YH + (jj + SR - dx)];
                    const float df = y0 - y1;
                    val = df * df;
                }
                bfrag[e] = (_Float16)val;
            }
            v8f cz = {};
            v8f dres = __builtin_amdgcn_wmma_f32_16x16x32_f16(
                false, avband, false, bfrag, (short)0, cz, false, false);
            if (half == 0) d1 = dres; else d2 = dres;
        }

        // Re-layout V (C/D layout -> A layout) through per-wave LDS.
        // d1 covers cols 0..15; d2 contributes only cols 16..19 (no overlap,
        // keeps the stores race-free / deterministic).
#pragma unroll
        for (int r = 0; r < 8; ++r) {
            const int m = r + (khalf << 3);
            myv[m * 20 + Mrow] = d1[r];
            if (Mrow >= 12) myv[m * 20 + Mrow + 4] = d2[r];
        }
        // same-wave DS ops are in-order; no barrier needed for private region

        // Stage 2: D[m][n] = sum_{u=0..4} V[m][n+u] = full 5x5 box sum.
        v16h afrag;
#pragma unroll
        for (int e = 0; e < 16; ++e) {
            const int v  = e >> 1, p = e & 1;
            const int ka = ((v & 3) << 1) + p + ((v >> 2) << 4) + (khalf << 3);
            const float val = (ka < 20) ? myv[Mrow * 20 + ka] : 0.0f;
            afrag[e] = (_Float16)val;
        }
        v8f cz2 = {};
        v8f dist2 = __builtin_amdgcn_wmma_f32_16x16x32_f16(
            false, afrag, false, bhband, (short)0, cz2, false, false);

        // Weights + weighted accumulation: raw v_sqrt_f32 + v_exp_f32.
        // dist2[r] >= 0 by construction, so sqrt needs no guard.
#pragma unroll
        for (int r = 0; r < 8; ++r) {
            const int m    = r + (khalf << 3);
            const float d  = __builtin_amdgcn_sqrtf(dist2[r]);
            const float w  = __builtin_amdgcn_exp2f(d * nlog2h);
            const int  ri  = (m - dy + SR) * RH + (Mrow - dx + SR);
            accr[r] += w * rgbl[0][ri];
            accg[r] += w * rgbl[1][ri];
            accb[r] += w * rgbl[2][ri];
            accw[r] += w;
        }
    }

    // ---- combine the 4 waves' partials and write the tile ----
#pragma unroll
    for (int r = 0; r < 8; ++r) {
        const int m   = r + (khalf << 3);
        const int pix = (m << 4) + Mrow;
        part[wave][0][pix] = accr[r];
        part[wave][1][pix] = accg[r];
        part[wave][2][pix] = accb[r];
        part[wave][3][pix] = accw[r];
    }
    __syncthreads();

    for (int pix = tid; pix < 256; pix += 32 * NW) {
        float sr = 0.0f, sg = 0.0f, sb = 0.0f, sw = 0.0f;
#pragma unroll
        for (int w = 0; w < NW; ++w) {
            sr += part[w][0][pix];
            sg += part[w][1][pix];
            sb += part[w][2][pix];
            sw += part[w][3][pix];
        }
        const int m  = pix >> 4, n = pix & 15;
        const int py = t0y + m,  px = t0x + n;
        const float inv = __builtin_amdgcn_rcpf(sw);   // sw >= 1 (zero-shift w = 1)
        out[0 * HW + py * IMG_W + px] = fminf(fmaxf(sr * inv, 0.0f), 1.0f);
        out[1 * HW + py * IMG_W + px] = fminf(fmaxf(sg * inv, 0.0f), 1.0f);
        out[2 * HW + py * IMG_W + px] = fminf(fmaxf(sb * inv, 0.0f), 1.0f);
    }
}

extern "C" void kernel_launch(void* const* d_in, const int* in_sizes, int n_in,
                              void* d_out, int out_size, void* d_ws, size_t ws_size,
                              hipStream_t stream) {
    (void)in_sizes; (void)n_in; (void)out_size; (void)d_ws; (void)ws_size;
    const float* rgb   = (const float*)d_in[0];   // [1,3,512,512] f32
    const float* sigma = (const float*)d_in[1];   // [1,1,1,1]     f32
    float* out = (float*)d_out;                   // [1,3,512,512] f32
    dim3 grid(IMG_W / 16, IMG_H / 16);            // one 16x16 tile per block
    nlm_wmma_kernel<<<grid, 32 * NW, 0, stream>>>(rgb, sigma, out);
}
